// FeatureConditionedIncidenceMask_45921790329369
// MI455X (gfx1250) — matrix operand order repaired and checked by reading, not compiled, not run on hardware
//
#include <hip/hip_runtime.h>
#include <math.h>

// Problem constants (match reference)
#define DD       64
#define HH       32
#define N_EDGES  200000
#define NBINS    4096
#define LIST_CAP 8192
#define W1T_PITCH 136   // halves per row (128 + 8 pad): 272B row -> 4-bank shift, 16B aligned

typedef __attribute__((ext_vector_type(16))) _Float16 v16h;
typedef __attribute__((ext_vector_type(8)))  float    v8f;

// ---------------- workspace layout (32-bit words) ----------------
static const long ESUM_W   = (long)N_EDGES * DD;   // edge sums -> edge feats (in place)
static const long ECNT_OFF = ESUM_W;               // edge counts (f32)
static const long EPS_OFF  = ECNT_OFF + N_EDGES;   // edge prob sums
static const long EHS_OFF  = EPS_OFF + N_EDGES;    // edge hard sums
static const long H1_OFF   = EHS_OFF + N_EDGES;    // histogram level 1
static const long H2_OFF   = H1_OFF + NBINS;       // histogram level 2
static const long PAR_OFF  = H2_OFF + NBINS;       // [0]=bin [1]=R [2]=sub [3]=R2 [4]=listcnt
static const long LIST_OFF = PAR_OFF + 16;         // boundary candidate indices
static const long TOTAL_W  = LIST_OFF + LIST_CAP;

// ---------------- kernels ----------------
__global__ void k_zero(unsigned* __restrict__ w, long n) {
  long i = (long)blockIdx.x * blockDim.x + threadIdx.x;
  if (i < n) w[i] = 0u;
}

// segment-sum of gathered node rows into edge accumulators (L2-resident atomics)
__global__ void k_scatter(const float* __restrict__ x, const int* __restrict__ V,
                          const int* __restrict__ E, float* __restrict__ esum,
                          float* __restrict__ ecnt, int nnz) {
  int i = blockIdx.x * blockDim.x + threadIdx.x;
  if (i >= nnz) return;
  int v = V[i], e = E[i];
  const float4* xs = (const float4*)(x + (size_t)v * DD);
  float* dst = esum + (size_t)e * DD;
#pragma unroll
  for (int q = 0; q < 16; ++q) {
    float4 t = xs[q];
    atomicAdd(dst + q * 4 + 0, t.x);
    atomicAdd(dst + q * 4 + 1, t.y);
    atomicAdd(dst + q * 4 + 2, t.z);
    atomicAdd(dst + q * 4 + 3, t.w);
  }
  atomicAdd(ecnt + e, 1.0f);
}

__global__ void k_divide(float* __restrict__ esum, const float* __restrict__ ecnt, long n) {
  long i = (long)blockIdx.x * blockDim.x + threadIdx.x;
  if (i >= n) return;
  float c = ecnt[i >> 6];            // DD == 64
  c = c < 1.0f ? 1.0f : c;
  esum[i] = esum[i] / c;
}

__device__ __forceinline__ int bin_of(float p) {
  int b = (int)(p * (float)NBINS);
  return b < 0 ? 0 : (b > NBINS - 1 ? NBINS - 1 : b);
}

// Fused gather + [xv|ef] @ W1 + relu + @W2 + sigmoid + histogram.
// One wave = 16 rows; K=128 via 4 chained v_wmma_f32_16x16x32_f16, N=32 via 2 tiles.
__global__ void __launch_bounds__(256)
k_mlp(const float* __restrict__ x, const float* __restrict__ ef,
      const int* __restrict__ V, const int* __restrict__ E,
      const float* __restrict__ W1, const float* __restrict__ b1,
      const float* __restrict__ W2, const float* __restrict__ b2,
      float* __restrict__ probs, unsigned* __restrict__ hist, int nnz) {
  // W1 transposed into LDS: w1t[n * W1T_PITCH + k] = W1[k*HH + n] (f16).
  // A lane's B operand = two 8-half runs 16 k apart -> two ds_load_b128.
  __shared__ _Float16 w1t[HH * W1T_PITCH];
  {
    int t = threadIdx.x;              // 256 threads, 16 elems each (4096 total)
    int n = t >> 3;                   // 0..31
    int k0 = (t & 7) * 16;            // 0,16,...,112
#pragma unroll
    for (int j = 0; j < 16; ++j)
      w1t[n * W1T_PITCH + k0 + j] = (_Float16)W1[(k0 + j) * HH + n];
  }
  __syncthreads();

  int lane = threadIdx.x & 31;
  int wave = threadIdx.x >> 5;
  int m    = lane & 15;
  int half = lane >> 4;
  long row0 = ((long)blockIdx.x * 8 + wave) * 16;
  long rowA = row0 + m;
  if (rowA >= nnz) rowA = nnz - 1;    // clamp loads; keep EXEC full for WMMA
  const float* px = x  + (size_t)V[rowA] * DD;
  const float* pe = ef + (size_t)E[rowA] * DD;

  v8f acc0 = {}; v8f acc1 = {};
#pragma unroll
  for (int c = 0; c < 4; ++c) {
    const float* src = ((c < 2) ? px : pe) + (c & 1) * 32 + half * 8;
    // ISA A-layout (16-bit A 16x32): lane-half holds K {base..base+7, base+16..base+23}
    float4 f0 = *(const float4*)(src + 0);
    float4 f1 = *(const float4*)(src + 4);
    float4 f2 = *(const float4*)(src + 16);
    float4 f3 = *(const float4*)(src + 20);
    v16h a;
    a[0]=(_Float16)f0.x; a[1]=(_Float16)f0.y; a[2]=(_Float16)f0.z; a[3]=(_Float16)f0.w;
    a[4]=(_Float16)f1.x; a[5]=(_Float16)f1.y; a[6]=(_Float16)f1.z; a[7]=(_Float16)f1.w;
    a[8]=(_Float16)f2.x; a[9]=(_Float16)f2.y; a[10]=(_Float16)f2.z; a[11]=(_Float16)f2.w;
    a[12]=(_Float16)f3.x; a[13]=(_Float16)f3.y; a[14]=(_Float16)f3.z; a[15]=(_Float16)f3.w;

    // B-layout matches A: lane-half holds K {8*half..8*half+7, 8*half+16..8*half+23}; col = m (+16)
    union { uint4 u[2]; v16h h; } cb0, cb1;
    const uint4* pB0 = (const uint4*)(w1t + m * W1T_PITCH + c * 32 + half * 8);
    const uint4* pB1 = (const uint4*)(w1t + (16 + m) * W1T_PITCH + c * 32 + half * 8);
    cb0.u[0] = pB0[0]; cb0.u[1] = pB0[2];
    cb1.u[0] = pB1[0]; cb1.u[1] = pB1[2];

    acc0 = __builtin_amdgcn_wmma_f32_16x16x32_f16(false, a, false, cb0.h, (short)0, acc0, false, false);
    acc1 = __builtin_amdgcn_wmma_f32_16x16x32_f16(false, a, false, cb1.h, (short)0, acc1, false, false);
  }

  float b1a = b1[m], b1b = b1[m + 16];
  float w2a = W2[m], w2b = W2[m + 16];
  float b2v = b2[0];

  float part[8];
#pragma unroll
  for (int r = 0; r < 8; ++r) {       // C layout: acc[r] = h[r + 8*half][n]
    float h0 = acc0[r] + b1a; h0 = h0 > 0.f ? h0 : 0.f;
    float h1 = acc1[r] + b1b; h1 = h1 > 0.f ? h1 : 0.f;
    part[r] = h0 * w2a + h1 * w2b;
  }
#pragma unroll
  for (int off = 1; off < 16; off <<= 1) {   // reduce over the 16 lanes of each half
#pragma unroll
    for (int r = 0; r < 8; ++r) part[r] += __shfl_xor(part[r], off, 32);
  }
  if (m < 8) {
    float z = 0.f;
#pragma unroll
    for (int r = 0; r < 8; ++r) if (m == r) z = part[r];
    z += b2v;
    float p = 1.0f / (1.0f + __expf(-z));
    long rr = row0 + half * 8 + m;
    if (rr < nnz) {
      probs[rr] = p;
      atomicAdd(&hist[bin_of(p)], 1u);       // fused level-1 histogram
    }
  }
}

__global__ void k_scan1(const unsigned* __restrict__ hist, unsigned* __restrict__ params, unsigned keep) {
  if (blockIdx.x || threadIdx.x) return;
  unsigned acc = 0;
  for (int b = NBINS - 1; b >= 0; --b) {
    unsigned prev = acc;
    acc += hist[b];
    if (acc >= keep) { params[0] = (unsigned)b; params[1] = keep - prev; return; }
  }
  params[0] = 0; params[1] = keep;
}

__global__ void k_hist2(const float* __restrict__ probs, const unsigned* __restrict__ params,
                        unsigned* __restrict__ hist2, int nnz) {
  int i = blockIdx.x * blockDim.x + threadIdx.x;
  if (i >= nnz) return;
  float p = probs[i];
  int b = bin_of(p);
  if (b != (int)params[0]) return;
  float f = p * (float)NBINS - (float)b;
  int s = (int)(f * (float)NBINS);
  s = s < 0 ? 0 : (s > NBINS - 1 ? NBINS - 1 : s);
  atomicAdd(&hist2[s], 1u);
}

__global__ void k_scan2(const unsigned* __restrict__ hist2, unsigned* __restrict__ params) {
  if (blockIdx.x || threadIdx.x) return;
  unsigned keep = params[1];
  unsigned acc = 0;
  for (int s = NBINS - 1; s >= 0; --s) {
    unsigned prev = acc;
    acc += hist2[s];
    if (acc >= keep) { params[2] = (unsigned)s; params[3] = keep - prev; return; }
  }
  params[2] = 0; params[3] = keep;
}

// hard/soft + edge accumulation fused (boundary elements deferred to k_rank)
__global__ void k_final(const float* __restrict__ probs, const unsigned* __restrict__ params,
                        unsigned* __restrict__ listcnt, unsigned* __restrict__ list_idx,
                        const int* __restrict__ E, float* __restrict__ eps,
                        float* __restrict__ ehs, float* __restrict__ soft,
                        float* __restrict__ hard, int nnz) {
  int i = blockIdx.x * blockDim.x + threadIdx.x;
  if (i >= nnz) return;
  float p = probs[i];
  int b  = bin_of(p);
  int bb = (int)params[0];
  int sb = (int)params[2];
  float h = 0.f;
  if (b > bb) h = 1.f;
  else if (b == bb) {
    float f = p * (float)NBINS - (float)b;
    int s = (int)(f * (float)NBINS);
    s = s < 0 ? 0 : (s > NBINS - 1 ? NBINS - 1 : s);
    if (s > sb) h = 1.f;
    else if (s == sb) {
      unsigned pos = atomicAdd(listcnt, 1u);
      if (pos < LIST_CAP) list_idx[pos] = (unsigned)i;
    }
  }
  hard[i] = h;
  soft[i] = h;                    // straight-through forward value == hard
  int e = E[i];
  atomicAdd(eps + e, p);
  atomicAdd(ehs + e, h);
}

// deterministic rank (value desc, index asc) among the few boundary candidates;
// patches hard/soft and the edge hard-sums for the promoted elements
__global__ void k_rank(const float* __restrict__ probs, const unsigned* __restrict__ params,
                       const unsigned* __restrict__ listcnt, const unsigned* __restrict__ list_idx,
                       const int* __restrict__ E, float* __restrict__ ehs,
                       float* __restrict__ soft, float* __restrict__ hard) {
  int n = (int)*listcnt; if (n > LIST_CAP) n = LIST_CAP;
  int R2 = (int)params[3];
  for (int t = threadIdx.x; t < n; t += blockDim.x) {
    unsigned idx = list_idx[t];
    float p = probs[idx];
    int rank = 0;
    for (int j = 0; j < n; ++j) {
      unsigned jd = list_idx[j];
      float q = probs[jd];
      if (q > p || (q == p && jd < idx)) rank++;
    }
    if (rank < R2) {
      hard[idx] = 1.f;
      soft[idx] = 1.f;
      atomicAdd(ehs + E[idx], 1.0f);
    }
  }
}

__global__ void k_eout(const float* __restrict__ eps, const float* __restrict__ ehs,
                       const float* __restrict__ ecnt, float* __restrict__ oep,
                       float* __restrict__ oes, float* __restrict__ oeh, int ne) {
  int e = blockIdx.x * blockDim.x + threadIdx.x;
  if (e >= ne) return;
  float c = ecnt[e]; c = c < 1.f ? 1.f : c;
  float hs = ehs[e];
  oep[e] = eps[e] / c;
  oes[e] = hs / c;                // soft == hard numerically
  oeh[e] = hs > 0.f ? 1.f : 0.f;
}

// ---------------- launcher ----------------
extern "C" void kernel_launch(void* const* d_in, const int* in_sizes, int n_in,
                              void* d_out, int out_size, void* d_ws, size_t ws_size,
                              hipStream_t stream) {
  const float* x  = (const float*)d_in[0];
  const int*   V  = (const int*)  d_in[1];
  const int*   E  = (const int*)  d_in[2];
  const float* W1 = (const float*)d_in[3];
  const float* b1 = (const float*)d_in[4];
  const float* W2 = (const float*)d_in[5];
  const float* b2 = (const float*)d_in[6];
  int nnz = in_sizes[1];
  unsigned keep = (unsigned)(nnz / 2); if (keep < 1u) keep = 1u;

  float* out      = (float*)d_out;
  float* oprobs   = out;
  float* osoft    = out + (size_t)nnz;
  float* ohard    = out + 2 * (size_t)nnz;
  float* oeprobs  = out + 3 * (size_t)nnz;
  float* oesoft   = oeprobs + N_EDGES;
  float* oehard   = oesoft + N_EDGES;

  unsigned* w       = (unsigned*)d_ws;
  float*    esum    = (float*)w;               // becomes edge_feats after k_divide
  float*    ecnt    = (float*)(w + ECNT_OFF);
  float*    eps     = (float*)(w + EPS_OFF);
  float*    ehs     = (float*)(w + EHS_OFF);
  unsigned* hist1   = w + H1_OFF;
  unsigned* hist2   = w + H2_OFF;
  unsigned* params  = w + PAR_OFF;
  unsigned* listcnt = params + 4;
  unsigned* listbuf = w + LIST_OFF;

  int nb = (nnz + 255) / 256;

  k_zero<<<(int)((TOTAL_W + 255) / 256), 256, 0, stream>>>(w, TOTAL_W);
  k_scatter<<<nb, 256, 0, stream>>>(x, V, E, esum, ecnt, nnz);
  long nd = (long)N_EDGES * DD;
  k_divide<<<(int)((nd + 255) / 256), 256, 0, stream>>>(esum, ecnt, nd);
  int mb = (nnz + 127) / 128;                  // 8 waves/block * 16 rows/wave
  k_mlp<<<mb, 256, 0, stream>>>(x, esum, V, E, W1, b1, W2, b2, oprobs, hist1, nnz);
  k_scan1<<<1, 1, 0, stream>>>(hist1, params, keep);
  k_hist2<<<nb, 256, 0, stream>>>(oprobs, params, hist2, nnz);
  k_scan2<<<1, 1, 0, stream>>>(hist2, params);
  k_final<<<nb, 256, 0, stream>>>(oprobs, params, listcnt, listbuf, E, eps, ehs, osoft, ohard, nnz);
  k_rank<<<1, 256, 0, stream>>>(oprobs, params, listcnt, listbuf, E, ehs, osoft, ohard);
  k_eout<<<(N_EDGES + 255) / 256, 256, 0, stream>>>(eps, ehs, ecnt, oeprobs, oesoft, oehard, N_EDGES);
}
